// ProbSparseAttention_13408887898230
// MI455X (gfx1250) — matrix operand order repaired and checked by reading, not compiled, not run on hardware
//
#include <hip/hip_runtime.h>
#include <hip/hip_bf16.h>

typedef __attribute__((ext_vector_type(16))) __bf16        v16bf;
typedef __attribute__((ext_vector_type(8)))  float         v8f;
typedef __attribute__((ext_vector_type(4)))  unsigned int  u32x4;

#define B_   8
#define S_   2048
#define D_   512
#define H_   8
#define DH_  64
#define MTOT (B_ * S_)   /* 16384 rows */
#define BH_  (B_ * H_)   /* 64 */
#define U_   40          /* 5 * ceil(ln(2048)) */

__device__ __forceinline__ unsigned short f2bf(float f) {
  unsigned int u = __builtin_bit_cast(unsigned int, f);
  u += 0x7FFFu + ((u >> 16) & 1u);               // round-to-nearest-even
  return (unsigned short)(u >> 16);
}
__device__ __forceinline__ float bf2f(unsigned short h) {
  unsigned int u = ((unsigned int)h) << 16;
  return __builtin_bit_cast(float, u);
}

union FragBF { v16bf v; u32x4 u[2]; };

// ---------------------------------------------------------------- converts
__global__ void cvt_bf16_kernel(const float* __restrict__ src,
                                unsigned short* __restrict__ dst, int n) {
  int i = blockIdx.x * blockDim.x + threadIdx.x;
  if (i < n) dst[i] = f2bf(src[i]);
}

// ------------------------------------------------- QKV projection (WMMA bf16)
// C[16384,512] = Xbf @ Wbf + bias, written head-split [B,H,S,Dh].
// OUTF32 = 1 -> float out (v), 0 -> bf16 out (q,k)
template <int OUTF32>
__global__ void gemm_proj_kernel(const unsigned short* __restrict__ A,
                                 const unsigned short* __restrict__ W,
                                 const float* __restrict__ bias,
                                 void* __restrict__ outp) {
  __shared__ alignas(16) unsigned short lA[128 * 32];     // row-major  [m][k]
  __shared__ alignas(16) unsigned short lB[64 * 32];      // col-major  [n][k]
  const int t = threadIdx.x;
  const int wave = t >> 5, lane = t & 31;
  const int m0 = blockIdx.y * 128;
  const int n0 = blockIdx.x * 64;
  const int g = lane >> 4;

  v8f acc[4];
#pragma unroll
  for (int nt = 0; nt < 4; ++nt)
#pragma unroll
    for (int j = 0; j < 8; ++j) acc[nt][j] = 0.0f;

  for (int kk = 0; kk < 16; ++kk) {
    const int k0 = kk * 32;
    { // stage A tile 128x32 (each thread 32 contiguous bytes)
      int r = t >> 1, half = t & 1;
      const u32x4* src = (const u32x4*)(A + (size_t)(m0 + r) * D_ + k0 + 16 * half);
      u32x4* dst = (u32x4*)(lA + r * 32 + 16 * half);
      dst[0] = src[0]; dst[1] = src[1];
    }
    { // stage B tile 32x64 transposed -> lB[n][k]
      int n = t & 63, kq = (t >> 6) * 8;
#pragma unroll
      for (int i = 0; i < 8; ++i)
        lB[n * 32 + kq + i] = W[(size_t)(k0 + kq + i) * D_ + n0 + n];
    }
    __syncthreads();

    FragBF a;  // A 16x32: lane holds row lane&15, K chunks [8g,8g+8),[8g+16,8g+24)
    {
      int m = 16 * wave + (lane & 15);
      int bk = g * 8;
      a.u[0] = *(const u32x4*)(lA + m * 32 + bk);
      a.u[1] = *(const u32x4*)(lA + m * 32 + bk + 16);
    }
    FragBF b[4]; // B 32x16: lane holds col lane&15, K chunks [8g,8g+8),[8g+16,8g+24)
#pragma unroll
    for (int nt = 0; nt < 4; ++nt) {
      int n = nt * 16 + (lane & 15);
      b[nt].u[0] = *(const u32x4*)(lB + n * 32 + g * 8);
      b[nt].u[1] = *(const u32x4*)(lB + n * 32 + g * 8 + 16);
    }
#pragma unroll
    for (int nt = 0; nt < 4; ++nt)
      acc[nt] = __builtin_amdgcn_wmma_f32_16x16x32_bf16(
          false, a.v, false, b[nt].v, (short)0, acc[nt], false, false);
    __syncthreads();
  }
  // epilogue: C lane layout m = j + 8g, n = lane&15; store head-split
#pragma unroll
  for (int nt = 0; nt < 4; ++nt) {
#pragma unroll
    for (int j = 0; j < 8; ++j) {
      int row = m0 + 16 * wave + j + 8 * g;
      int col = n0 + nt * 16 + (lane & 15);
      float val = acc[nt][j] + bias[col];
      int b2 = row >> 11, s = row & 2047;
      int h = col >> 6, d = col & 63;
      size_t idx = ((size_t)(b2 * H_ + h) * S_ + s) * DH_ + d;
      if (OUTF32) ((float*)outp)[idx] = val;
      else        ((unsigned short*)outp)[idx] = f2bf(val);
    }
  }
}

// ------------------------------------------------------------- k/v means
__global__ void means_kernel(const unsigned short* __restrict__ kb,
                             const float* __restrict__ vf,
                             float* __restrict__ kmean, float* __restrict__ vmean) {
  int bh = blockIdx.x, d = threadIdx.x;
  size_t base = (size_t)bh * S_ * DH_ + d;
  float ks = 0.f, vs = 0.f;
  for (int s = 0; s < S_; ++s) {
    ks += bf2f(kb[base + (size_t)s * DH_]);
    vs += vf[base + (size_t)s * DH_];
  }
  kmean[bh * DH_ + d] = ks * (1.0f / S_);
  vmean[bh * DH_ + d] = vs * (1.0f / S_);
}

// ---------------------- row-max of Q@K^T via WMMA, M = (max - q.kmean)/sqrt(Dh)
// One wave holds 64 Q rows (4 x 16-row subtiles, 64 VGPRs of A fragments) so
// each streamed K tile (4 clause-merged global b128 loads) feeds 8 WMMAs.
// __launch_bounds__(256,1): favor VGPRs over occupancy -> no scratch spills.
__global__ void __launch_bounds__(256, 1)
rowmax_kernel(const unsigned short* __restrict__ qb,
              const unsigned short* __restrict__ kb,
              const float* __restrict__ kmean,
              float* __restrict__ Mout) {
  int gw   = blockIdx.x * 8 + (threadIdx.x >> 5);   // 2048 waves
  int lane = threadIdx.x & 31;
  int bh = gw >> 5, qblk = gw & 31;                 // 64 bh x 32 blocks of 64 rows
  size_t base = (size_t)bh * S_ * DH_;
  int g = lane >> 4;

  FragBF a0[4], a1[4];  // Q rows: two K=32 fragments per 16-row subtile
#pragma unroll
  for (int p = 0; p < 4; ++p) {
    int qrow = qblk * 64 + p * 16 + (lane & 15);
    const u32x4* qp = (const u32x4*)(qb + base + (size_t)qrow * DH_);
    a0[p].u[0] = qp[g];     a0[p].u[1] = qp[g + 2];
    a1[p].u[0] = qp[g + 4]; a1[p].u[1] = qp[g + 6];
  }
  v8f rmax[4];
#pragma unroll
  for (int p = 0; p < 4; ++p)
#pragma unroll
    for (int j = 0; j < 8; ++j) rmax[p][j] = -__builtin_inff();

#pragma unroll 1
  for (int kt = 0; kt < 128; ++kt) {
    int krow = kt * 16 + (lane & 15);
    const u32x4* kp = (const u32x4*)(kb + base + (size_t)krow * DH_);
    FragBF b0, b1;  // rows of K are exactly the n-major B fragment
    b0.u[0] = kp[g];     b0.u[1] = kp[g + 2];
    b1.u[0] = kp[4 + g]; b1.u[1] = kp[6 + g];
#pragma unroll
    for (int p = 0; p < 4; ++p) {
      v8f c;
#pragma unroll
      for (int j = 0; j < 8; ++j) c[j] = 0.0f;
      c = __builtin_amdgcn_wmma_f32_16x16x32_bf16(false, a0[p].v, false, b0.v, (short)0, c, false, false);
      c = __builtin_amdgcn_wmma_f32_16x16x32_bf16(false, a1[p].v, false, b1.v, (short)0, c, false, false);
#pragma unroll
      for (int j = 0; j < 8; ++j) rmax[p][j] = fmaxf(rmax[p][j], c[j]);
    }
  }
  // max over N lives across the 16 lanes of each half-wave group
#pragma unroll
  for (int mask = 1; mask < 16; mask <<= 1)
#pragma unroll
    for (int p = 0; p < 4; ++p)
#pragma unroll
      for (int j = 0; j < 8; ++j)
        rmax[p][j] = fmaxf(rmax[p][j], __shfl_xor(rmax[p][j], mask, 32));

  if ((lane & 15) < 8) {
    int j = lane & 15;
    const float* km = kmean + bh * DH_;
#pragma unroll 1
    for (int p = 0; p < 4; ++p) {
      int row = qblk * 64 + p * 16 + j + 8 * g;
      const unsigned short* qrow = qb + base + (size_t)row * DH_;
      float dot = 0.f;
#pragma unroll 8
      for (int d = 0; d < DH_; ++d) dot += bf2f(qrow[d]) * km[d];
      Mout[(size_t)bh * S_ + row] = (rmax[p][j] - dot) * 0.125f;
    }
  }
}

// ------------------------------------------------------- top-40 per (b,h)
__global__ void topk_kernel(const float* __restrict__ M, int* __restrict__ topidx) {
  __shared__ float vals[S_];
  __shared__ float rv[256];
  __shared__ int   ri[256];
  int bh = blockIdx.x, t = threadIdx.x;
  for (int i = t; i < S_; i += 256) vals[i] = M[(size_t)bh * S_ + i];
  __syncthreads();
  for (int it = 0; it < U_; ++it) {
    float best = -__builtin_inff(); int bidx = 0;
    for (int i = t; i < S_; i += 256)
      if (vals[i] > best) { best = vals[i]; bidx = i; }
    rv[t] = best; ri[t] = bidx;
    __syncthreads();
    for (int s = 128; s > 0; s >>= 1) {
      if (t < s) {
        if (rv[t + s] > rv[t] || (rv[t + s] == rv[t] && ri[t + s] < ri[t])) {
          rv[t] = rv[t + s]; ri[t] = ri[t + s];
        }
      }
      __syncthreads();
    }
    if (t == 0) { topidx[bh * U_ + it] = ri[0]; vals[ri[0]] = -__builtin_inff(); }
    __syncthreads();
  }
}

// ---------------------------------------------- out_heads = broadcast vmean
__global__ void fill_mean_kernel(const float* __restrict__ vmean,
                                 float* __restrict__ outh) {
  size_t idx = (size_t)blockIdx.x * 256 + threadIdx.x;
  if (idx < (size_t)BH_ * S_ * DH_) {
    int d  = (int)(idx & 63);
    int bh = (int)(idx >> 17);   // S_*DH_ = 2^17
    outh[idx] = vmean[bh * DH_ + d];
  }
}

// ------------------------------- dense attention for the 40 selected queries
__global__ void topattn_kernel(const unsigned short* __restrict__ qb,
                               const unsigned short* __restrict__ kb,
                               const float* __restrict__ vf,
                               const int* __restrict__ topidx,
                               float* __restrict__ outh) {
  __shared__ float P[4][S_];
  __shared__ float Q[4][DH_];
  int wave = threadIdx.x >> 5, lane = threadIdx.x & 31;
  int gw = blockIdx.x * 4 + wave;
  int bh = gw / U_, i = gw % U_;
  int row = topidx[bh * U_ + i];
  size_t base = (size_t)bh * S_ * DH_;

  const unsigned short* qrow = qb + base + (size_t)row * DH_;
  Q[wave][lane]      = bf2f(qrow[lane]);
  Q[wave][lane + 32] = bf2f(qrow[lane + 32]);
  __syncthreads();

  float lmax = -__builtin_inff();
  for (int s = lane; s < S_; s += 32) {
    const unsigned short* krow = kb + base + (size_t)s * DH_;
    float dot = 0.f;
#pragma unroll 8
    for (int d = 0; d < DH_; ++d) dot += Q[wave][d] * bf2f(krow[d]);
    float lg = dot * 0.125f;
    P[wave][s] = lg;
    lmax = fmaxf(lmax, lg);
  }
  for (int mask = 1; mask < 32; mask <<= 1) lmax = fmaxf(lmax, __shfl_xor(lmax, mask, 32));
  __syncthreads();

  float lsum = 0.f;
  for (int s = lane; s < S_; s += 32) {
    float p = __expf(P[wave][s] - lmax);
    P[wave][s] = p;
    lsum += p;
  }
  for (int mask = 1; mask < 32; mask <<= 1) lsum += __shfl_xor(lsum, mask, 32);
  __syncthreads();

  float acc0 = 0.f, acc1 = 0.f;
  const float* vb = vf + base;
  for (int s = 0; s < S_; ++s) {
    float p = P[wave][s];
    acc0 += p * vb[(size_t)s * DH_ + lane];
    acc1 += p * vb[(size_t)s * DH_ + lane + 32];
  }
  float rinv = 1.0f / lsum;
  outh[base + (size_t)row * DH_ + lane]      = acc0 * rinv;
  outh[base + (size_t)row * DH_ + lane + 32] = acc1 * rinv;
}

// ------------------------------------------- output projection (WMMA bf16)
__global__ void gemm_oproj_kernel(const float* __restrict__ OH,
                                  const unsigned short* __restrict__ W,
                                  const float* __restrict__ bias,
                                  float* __restrict__ out) {
  __shared__ alignas(16) unsigned short lA[128 * 32];
  __shared__ alignas(16) unsigned short lB[64 * 32];
  const int t = threadIdx.x;
  const int wave = t >> 5, lane = t & 31;
  const int m0 = blockIdx.y * 128;
  const int n0 = blockIdx.x * 64;
  const int g = lane >> 4;

  v8f acc[4];
#pragma unroll
  for (int nt = 0; nt < 4; ++nt)
#pragma unroll
    for (int j = 0; j < 8; ++j) acc[nt][j] = 0.0f;

  for (int kk = 0; kk < 16; ++kk) {
    const int k0 = kk * 32;
    { // gather head-split f32 -> bf16 A tile (16 consecutive d stay in one head)
      int r = t >> 1, half = t & 1;
      int row = m0 + r;
      int b2 = row >> 11, s = row & 2047;
      int cbase = k0 + 16 * half;
      int h = cbase >> 6, d0 = cbase & 63;
      const float* src = OH + ((size_t)(b2 * H_ + h) * S_ + s) * DH_ + d0;
      unsigned short* dst = lA + r * 32 + 16 * half;
#pragma unroll
      for (int i2 = 0; i2 < 16; ++i2) dst[i2] = f2bf(src[i2]);
    }
    {
      int n = t & 63, kq = (t >> 6) * 8;
#pragma unroll
      for (int i2 = 0; i2 < 8; ++i2)
        lB[n * 32 + kq + i2] = W[(size_t)(k0 + kq + i2) * D_ + n0 + n];
    }
    __syncthreads();

    FragBF a;
    {
      int m = 16 * wave + (lane & 15);
      int bk = g * 8;
      a.u[0] = *(const u32x4*)(lA + m * 32 + bk);
      a.u[1] = *(const u32x4*)(lA + m * 32 + bk + 16);
    }
    FragBF b[4];
#pragma unroll
    for (int nt = 0; nt < 4; ++nt) {
      int n = nt * 16 + (lane & 15);
      b[nt].u[0] = *(const u32x4*)(lB + n * 32 + g * 8);
      b[nt].u[1] = *(const u32x4*)(lB + n * 32 + g * 8 + 16);
    }
#pragma unroll
    for (int nt = 0; nt < 4; ++nt)
      acc[nt] = __builtin_amdgcn_wmma_f32_16x16x32_bf16(
          false, a.v, false, b[nt].v, (short)0, acc[nt], false, false);
    __syncthreads();
  }
#pragma unroll
  for (int nt = 0; nt < 4; ++nt) {
#pragma unroll
    for (int j = 0; j < 8; ++j) {
      int row = m0 + 16 * wave + j + 8 * g;
      int col = n0 + nt * 16 + (lane & 15);
      out[(size_t)row * D_ + col] = acc[nt][j] + bias[col];
    }
  }
}

// =========================================================================
extern "C" void kernel_launch(void* const* d_in, const int* in_sizes, int n_in,
                              void* d_out, int out_size, void* d_ws, size_t ws_size,
                              hipStream_t stream) {
  (void)in_sizes; (void)n_in; (void)out_size; (void)ws_size;
  const float* x  = (const float*)d_in[0];
  const float* wq = (const float*)d_in[1];
  const float* bq = (const float*)d_in[2];
  const float* wk = (const float*)d_in[3];
  const float* bk = (const float*)d_in[4];
  const float* wv = (const float*)d_in[5];
  const float* bv = (const float*)d_in[6];
  const float* wo = (const float*)d_in[7];
  const float* bo = (const float*)d_in[8];
  float* out = (float*)d_out;

  char* ws = (char*)d_ws;
  size_t off = 0;
  auto alloc = [&](size_t bytes) -> void* {
    void* p = ws + off;
    off += (bytes + 255) & ~(size_t)255;
    return p;
  };
  unsigned short* xb    = (unsigned short*)alloc((size_t)MTOT * D_ * 2);   // 16 MB
  unsigned short* wqb   = (unsigned short*)alloc((size_t)D_ * D_ * 2);
  unsigned short* wkb   = (unsigned short*)alloc((size_t)D_ * D_ * 2);
  unsigned short* wvb   = (unsigned short*)alloc((size_t)D_ * D_ * 2);
  unsigned short* wob   = (unsigned short*)alloc((size_t)D_ * D_ * 2);
  unsigned short* qbuf  = (unsigned short*)alloc((size_t)MTOT * D_ * 2);   // [B,H,S,Dh]
  unsigned short* kbuf  = (unsigned short*)alloc((size_t)MTOT * D_ * 2);
  float*          vbuf  = (float*)alloc((size_t)MTOT * D_ * 4);            // [B,H,S,Dh]
  float*          kmean = (float*)alloc((size_t)BH_ * DH_ * 4);
  float*          vmean = (float*)alloc((size_t)BH_ * DH_ * 4);
  float*          Mbuf  = (float*)alloc((size_t)BH_ * S_ * 4);
  int*            tidx  = (int*)alloc((size_t)BH_ * U_ * 4);
  float*          oh    = (float*)alloc((size_t)MTOT * D_ * 4);            // [B,H,S,Dh]

  const int nx = MTOT * D_;      // 8,388,608
  const int nw = D_ * D_;        // 262,144
  cvt_bf16_kernel<<<(nx + 255) / 256, 256, 0, stream>>>(x,  xb,  nx);
  cvt_bf16_kernel<<<(nw + 255) / 256, 256, 0, stream>>>(wq, wqb, nw);
  cvt_bf16_kernel<<<(nw + 255) / 256, 256, 0, stream>>>(wk, wkb, nw);
  cvt_bf16_kernel<<<(nw + 255) / 256, 256, 0, stream>>>(wv, wvb, nw);
  cvt_bf16_kernel<<<(nw + 255) / 256, 256, 0, stream>>>(wo, wob, nw);

  dim3 gg(D_ / 64, MTOT / 128);  // (8, 128)
  gemm_proj_kernel<0><<<gg, 256, 0, stream>>>(xb, wqb, bq, (void*)qbuf);
  gemm_proj_kernel<0><<<gg, 256, 0, stream>>>(xb, wkb, bk, (void*)kbuf);
  gemm_proj_kernel<1><<<gg, 256, 0, stream>>>(xb, wvb, bv, (void*)vbuf);

  means_kernel<<<BH_, DH_, 0, stream>>>(kbuf, vbuf, kmean, vmean);
  rowmax_kernel<<<256, 256, 0, stream>>>(qbuf, kbuf, kmean, Mbuf);
  topk_kernel<<<BH_, 256, 0, stream>>>(Mbuf, tidx);

  fill_mean_kernel<<<(nx + 255) / 256, 256, 0, stream>>>(vmean, oh);
  topattn_kernel<<<(BH_ * U_) / 4, 128, 0, stream>>>(qbuf, kbuf, vbuf, tidx, oh);

  gemm_oproj_kernel<<<gg, 256, 0, stream>>>(oh, wob, bo, out);
}
